// _3DIS_31860067402336
// MI455X (gfx1250) — matrix operand order repaired and checked by reading, not compiled, not run on hardware
//
#include <hip/hip_runtime.h>
#include <hip/hip_bf16.h>
#include <math.h>

// ---- problem constants (from reference) ----
#define BATCH   131072
#define N_TOOL  6
#define N_VERB  10
#define N_TARG  15
#define N_TRIP  100
#define N_COLS  31          // 6 + 10 + 15
#define STATS_BLOCKS 256

typedef __attribute__((ext_vector_type(16))) _Float16 v16h;
typedef __attribute__((ext_vector_type(8)))  float    v8f;

union Frag16 { v16h h; unsigned u[8]; };

__device__ __forceinline__ unsigned pack2f16(float a, float b) {
    union { _Float16 h[2]; unsigned u; } r;
    r.h[0] = (_Float16)a;
    r.h[1] = (_Float16)b;
    return r.u;
}

// ============================================================================
// Kernel 1: per-block partial sums of y and y^2 for the 31 branch columns.
// Deterministic: fixed-order shfl tree + fixed-order LDS fold, no atomics.
// ============================================================================
__global__ __launch_bounds__(256) void k_stats(
    const float* __restrict__ tl, const float* __restrict__ vl,
    const float* __restrict__ gl,
    const float* __restrict__ alpha, const float* __restrict__ beta,
    const float* __restrict__ gamma,
    float* __restrict__ partial)   // [STATS_BLOCKS][62]
{
    __shared__ float sA[36], sB[100], sG[225];
    __shared__ float wred[8][62];
    const int tid = threadIdx.x;
    for (int i = tid; i < 36;  i += 256) sA[i] = alpha[i];
    for (int i = tid; i < 100; i += 256) sB[i] = beta[i];
    for (int i = tid; i < 225; i += 256) sG[i] = gamma[i];
    __syncthreads();

    float s[N_COLS], q[N_COLS];
#pragma unroll
    for (int c = 0; c < N_COLS; ++c) { s[c] = 0.f; q[c] = 0.f; }

    const int stride = gridDim.x * 256;
    for (int row = blockIdx.x * 256 + tid; row < BATCH; row += stride) {
        float x[15];
#pragma unroll
        for (int k = 0; k < 6; ++k) x[k] = tl[row * 6 + k];
#pragma unroll
        for (int j = 0; j < 6; ++j) {
            float y = 0.f;
#pragma unroll
            for (int k = 0; k < 6; ++k) y = fmaf(x[k], sA[k * 6 + j], y);
            s[j] += y; q[j] = fmaf(y, y, q[j]);
        }
#pragma unroll
        for (int k = 0; k < 10; ++k) x[k] = vl[row * 10 + k];
#pragma unroll
        for (int j = 0; j < 10; ++j) {
            float y = 0.f;
#pragma unroll
            for (int k = 0; k < 10; ++k) y = fmaf(x[k], sB[k * 10 + j], y);
            s[6 + j] += y; q[6 + j] = fmaf(y, y, q[6 + j]);
        }
#pragma unroll
        for (int k = 0; k < 15; ++k) x[k] = gl[row * 15 + k];
#pragma unroll
        for (int j = 0; j < 15; ++j) {
            float y = 0.f;
#pragma unroll
            for (int k = 0; k < 15; ++k) y = fmaf(x[k], sG[k * 15 + j], y);
            s[16 + j] += y; q[16 + j] = fmaf(y, y, q[16 + j]);
        }
    }

    // wave32 deterministic tree reduction
#pragma unroll
    for (int c = 0; c < N_COLS; ++c) {
        float a = s[c], b = q[c];
#pragma unroll
        for (int off = 16; off > 0; off >>= 1) {
            a += __shfl_xor(a, off, 32);
            b += __shfl_xor(b, off, 32);
        }
        s[c] = a; q[c] = b;
    }
    const int wv = tid >> 5, ln = tid & 31;
    if (ln == 0) {
#pragma unroll
        for (int c = 0; c < N_COLS; ++c) {
            wred[wv][c] = s[c];
            wred[wv][N_COLS + c] = q[c];
        }
    }
    __syncthreads();
    if (tid < 2 * N_COLS) {
        float acc = 0.f;
#pragma unroll
        for (int i = 0; i < 8; ++i) acc += wred[i][tid];
        partial[blockIdx.x * (2 * N_COLS) + tid] = acc;
    }
}

// ============================================================================
// Kernel 2: fold block partials -> 62 final sums (fixed order, deterministic)
// ============================================================================
__global__ __launch_bounds__(64) void k_reduce(
    const float* __restrict__ partial, float* __restrict__ fin)
{
    const int c = threadIdx.x;
    if (c < 2 * N_COLS) {
        float acc = 0.f;
        for (int b = 0; b < STATS_BLOCKS; ++b)
            acc += partial[b * (2 * N_COLS) + c];
        fin[c] = acc;
    }
}

// ============================================================================
// Kernel 3: fused branch+BN+ELU+gather, then [B,100]x[100,100]^T via WMMA.
// 128 threads = 4 waves; each wave owns 32 rows (two 16x128 f16 A tiles).
// ============================================================================
__global__ __launch_bounds__(128) void k_main(
    const float* __restrict__ tl, const float* __restrict__ vl,
    const float* __restrict__ gl,
    const float* __restrict__ alpha, const float* __restrict__ beta,
    const float* __restrict__ gamma,
    const float* __restrict__ bn1w, const float* __restrict__ bn1b,
    const float* __restrict__ bn2w, const float* __restrict__ bn2b,
    const float* __restrict__ bn3w, const float* __restrict__ bn3b,
    const float* __restrict__ mlpw, const float* __restrict__ mlpb,
    const int*   __restrict__ vpos,
    const float* __restrict__ stats,   // 62 finals: sum[31], sumsq[31]
    float* __restrict__ out)
{
    __shared__ unsigned As[4][32][64];    // 4 waves x 32 rows x 128 f16 (pairs)   32 KB
    __shared__ unsigned Wp[112][64];      // B^T padded: [n][k] f16 pairs          28 KB
    __shared__ float sA[36], sB[100], sG[225];
    __shared__ float sScale[N_COLS], sShift[N_COLS], sBias[112];
    __shared__ unsigned sIdx[N_TRIP];     // packed LDS offsets (i | (6+v)<<8 | (16+t)<<16)

    const int tid = threadIdx.x;

    for (int i = tid; i < 36;  i += 128) sA[i] = alpha[i];
    for (int i = tid; i < 100; i += 128) sB[i] = beta[i];
    for (int i = tid; i < 225; i += 128) sG[i] = gamma[i];
    for (int i = tid; i < N_TRIP; i += 128) {
        int p   = vpos[i];
        int ii  = p / (N_VERB * N_TARG);
        int rem = p - ii * (N_VERB * N_TARG);
        int vv  = rem / N_TARG;
        int tt  = rem - vv * N_TARG;
        sIdx[i] = (unsigned)ii | ((unsigned)(6 + vv) << 8) | ((unsigned)(16 + tt) << 16);
    }
    for (int i = tid; i < 112; i += 128) sBias[i] = (i < 100) ? mlpb[i] : 0.f;
    for (int c = tid; c < N_COLS; c += 128) {
        const float inv = 1.f / (float)BATCH;
        float mean = stats[c] * inv;
        float var  = stats[N_COLS + c] * inv - mean * mean;
        float rstd = rsqrtf(var + 1e-5f);
        float g, b2;
        if (c < 6)       { g = bn1w[c];      b2 = bn1b[c]; }
        else if (c < 16) { g = bn2w[c - 6];  b2 = bn2b[c - 6]; }
        else             { g = bn3w[c - 16]; b2 = bn3b[c - 16]; }
        sScale[c] = g * rstd;
        sShift[c] = b2 - mean * g * rstd;
    }
    // stage mlp_w as f16, n-major, zero-padded to 112x128
    for (int i = tid; i < 112 * 64; i += 128) {
        int n = i >> 6, kp = i & 63, k = kp * 2;
        float a  = (n < 100 && k     < 100) ? mlpw[n * 100 + k]     : 0.f;
        float b2 = (n < 100 && k + 1 < 100) ? mlpw[n * 100 + k + 1] : 0.f;
        Wp[n][kp] = pack2f16(a, b2);
    }
    __syncthreads();

    const int w = tid >> 5, l = tid & 31;
    const int rowBase = blockIdx.x * 128;

    // ---- per-lane row: branches -> BN -> ELU -> gather -> f16 A row ----
    {
        const int row = rowBase + w * 32 + l;
        unsigned* myrow = &As[w][l][0];
        float x[15];
#pragma unroll
        for (int k = 0; k < 6; ++k) x[k] = tl[row * 6 + k];
#pragma unroll
        for (int j = 0; j < 6; ++j) {
            float y = 0.f;
#pragma unroll
            for (int k = 0; k < 6; ++k) y = fmaf(x[k], sA[k * 6 + j], y);
            y = fmaf(y, sScale[j], sShift[j]);
            myrow[j] = __float_as_uint(y > 0.f ? y : expm1f(y));
        }
#pragma unroll
        for (int k = 0; k < 10; ++k) x[k] = vl[row * 10 + k];
#pragma unroll
        for (int j = 0; j < 10; ++j) {
            float y = 0.f;
#pragma unroll
            for (int k = 0; k < 10; ++k) y = fmaf(x[k], sB[k * 10 + j], y);
            y = fmaf(y, sScale[6 + j], sShift[6 + j]);
            myrow[6 + j] = __float_as_uint(y > 0.f ? y : expm1f(y));
        }
#pragma unroll
        for (int k = 0; k < 15; ++k) x[k] = gl[row * 15 + k];
#pragma unroll
        for (int j = 0; j < 15; ++j) {
            float y = 0.f;
#pragma unroll
            for (int k = 0; k < 15; ++k) y = fmaf(x[k], sG[k * 15 + j], y);
            y = fmaf(y, sScale[16 + j], sShift[16 + j]);
            myrow[16 + j] = __float_as_uint(y > 0.f ? y : expm1f(y));
        }
        float validv[N_TRIP];
#pragma unroll
        for (int p = 0; p < N_TRIP; ++p) {
            unsigned e = sIdx[p];
            float a = __uint_as_float(myrow[e & 0xFFu]);
            float b = __uint_as_float(myrow[(e >> 8) & 0xFFu]);
            float c = __uint_as_float(myrow[(e >> 16) & 0xFFu]);
            validv[p] = a * b * c;
        }
#pragma unroll
        for (int pp = 0; pp < 50; ++pp)
            myrow[pp] = pack2f16(validv[2 * pp], validv[2 * pp + 1]);
#pragma unroll
        for (int pp = 50; pp < 64; ++pp) myrow[pp] = 0u;
    }
    __syncthreads();

    // ---- WMMA GEMM: D[32x112] = A[32x128] * W^T[128x112] + bias ----
    const int h = l >> 4, m = l & 15;

    // A fragments, contiguous k order (matches B):
    // K = h*16 + 2v .. +1, M = m
    Frag16 Af[2][4];
#pragma unroll
    for (int t = 0; t < 2; ++t)
#pragma unroll
        for (int kk = 0; kk < 4; ++kk)
#pragma unroll
            for (int v = 0; v < 8; ++v)
                Af[t][kk].u[v] =
                    As[w][t * 16 + m][kk * 16 + h * 8 + v];

#pragma unroll
    for (int nt = 0; nt < 7; ++nt) {
        const int n = nt * 16 + m;
        // B fragment, contiguous k order (matches A): K = h*16 + 2v .. +1, N = m
        Frag16 Bf[4];
#pragma unroll
        for (int kk = 0; kk < 4; ++kk)
#pragma unroll
            for (int v = 0; v < 8; ++v)
                Bf[kk].u[v] = Wp[n][kk * 16 + h * 8 + v];

        v8f acc0 = {};
        v8f acc1 = {};
#pragma unroll
        for (int kk = 0; kk < 4; ++kk) {
            acc0 = __builtin_amdgcn_wmma_f32_16x16x32_f16(
                false, Af[0][kk].h, false, Bf[kk].h, (short)0, acc0, false, false);
            acc1 = __builtin_amdgcn_wmma_f32_16x16x32_f16(
                false, Af[1][kk].h, false, Bf[kk].h, (short)0, acc1, false, false);
        }
        if (n < 100) {
            const float bias = sBias[n];
            const int r0 = rowBase + w * 32 + h * 8;   // C/D layout: M = r + 8*h
#pragma unroll
            for (int r = 0; r < 8; ++r) {
                out[(r0 + r) * 100 + n]      = acc0[r] + bias;
                out[(r0 + 16 + r) * 100 + n] = acc1[r] + bias;
            }
        }
    }
}

// ============================================================================
extern "C" void kernel_launch(void* const* d_in, const int* in_sizes, int n_in,
                              void* d_out, int out_size, void* d_ws, size_t ws_size,
                              hipStream_t stream) {
    const float* tl    = (const float*)d_in[0];
    const float* vl    = (const float*)d_in[1];
    const float* gl    = (const float*)d_in[2];
    const float* alpha = (const float*)d_in[3];
    const float* beta  = (const float*)d_in[4];
    const float* gamma = (const float*)d_in[5];
    const float* bn1w  = (const float*)d_in[6];
    const float* bn1b  = (const float*)d_in[7];
    const float* bn2w  = (const float*)d_in[8];
    const float* bn2b  = (const float*)d_in[9];
    const float* bn3w  = (const float*)d_in[10];
    const float* bn3b  = (const float*)d_in[11];
    const float* mlpw  = (const float*)d_in[12];
    const float* mlpb  = (const float*)d_in[13];
    const int*   vpos  = (const int*)d_in[14];

    float* ws      = (float*)d_ws;
    float* partial = ws;                              // STATS_BLOCKS * 62
    float* fin     = ws + STATS_BLOCKS * (2 * N_COLS); // 62 finals

    k_stats<<<STATS_BLOCKS, 256, 0, stream>>>(tl, vl, gl, alpha, beta, gamma, partial);
    k_reduce<<<1, 64, 0, stream>>>(partial, fin);
    k_main<<<BATCH / 128, 128, 0, stream>>>(tl, vl, gl, alpha, beta, gamma,
                                            bn1w, bn1b, bn2w, bn2b, bn3w, bn3b,
                                            mlpw, mlpb, vpos, fin, (float*)d_out);
}
